// TGN_3959959847021
// MI455X (gfx1250) — matrix operand order repaired and checked by reading, not compiled, not run on hardware
//
#include <hip/hip_runtime.h>

#define LDM 264    // bf16 LDS row stride (halves) for 256-wide panels
#define LDH1 136   // bf16 LDS row stride for 128-wide panel

// packed weight offsets in halves inside d_ws
#define O_W1  0        // 16x8  frags  (K=512, N=128)
#define O_W2  65536    // 4x16  frags  (K=128, N=256)
#define O_WIH 98304    // 8x48  frags  (K=256, N=768)  (w_ih transposed)
#define O_WHH 294912   // 8x48  frags
#define O_P1  491520   // 16x8  frags  (K=512, N=128)

typedef __attribute__((ext_vector_type(16))) __bf16 v16bf;
typedef __attribute__((ext_vector_type(8)))  float  v8f;

union Frag {
  v16bf v;
  __bf16 s[16];
  uint4 q[2];
};

union BF4 {
  __bf16 h[4];
  uint2 u;
};

struct AccPair { v8f a[2]; };

__device__ __forceinline__ uint2 pack4bf(float4 v) {
  BF4 b;
  b.h[0] = (__bf16)v.x;
  b.h[1] = (__bf16)v.y;
  b.h[2] = (__bf16)v.z;
  b.h[3] = (__bf16)v.w;
  return b.u;
}

__device__ __forceinline__ v8f wmma_bf16(v16bf a, v16bf b, v8f c) {
  return __builtin_amdgcn_wmma_f32_16x16x32_bf16(false, a, false, b, (short)0, c, false, false);
}

// A fragment from bf16 LDS panel (two ds_load_b128)
// lane<16: row=lane, K = kb+0..7, kb+16..23 ; lane>=16: K = kb+8..15, kb+24..31
__device__ __forceinline__ v16bf loadA_bf(const __bf16* base, int ld, int kbase, int lane) {
  int m = lane & 15, hh = lane >> 4;
  const __bf16* p = base + m * ld + kbase + hh * 8;
  Frag f;
  f.q[0] = *(const uint4*)(p);
  f.q[1] = *(const uint4*)(p + 16);
  return f.v;
}

// B fragment (32x16 bf16), fragment-major packed in ws: 512 halves per frag, lane-major
__device__ __forceinline__ v16bf loadB(const __bf16* Bp, int frag, int lane) {
  const uint4* q = (const uint4*)(Bp + (long)frag * 512) + lane * 2;
  Frag f;
  f.q[0] = q[0];
  f.q[1] = q[1];
  return f.v;
}

// ---- weight packer: B[k][n] -> frag-major bf16 ----
__global__ void tgn_pack_kernel(const float* __restrict__ W, __bf16* __restrict__ dstp,
                                int KT, int NT, int ldw, int trans) {
  int idx = blockIdx.x * blockDim.x + threadIdx.x;
  int total = KT * NT * 512;
  if (idx >= total) return;
  int frag = idx >> 9;
  int rem  = idx & 511;
  int lane = rem >> 4;
  int jj   = rem & 15;
  int ntile = frag / KT;
  int ktile = frag % KT;
  int kk = ktile * 32 + (jj >> 3) * 16 + (lane >> 4) * 8 + (jj & 7);   // B lane layout: lane>>4 selects 8-wide K group within each 16-half
  int nn = ntile * 16 + (lane & 15);
  float v = trans ? W[(long)nn * ldw + kk] : W[(long)kk * ldw + nn];
  dstp[idx] = (__bf16)v;
}

// dual-row-tile GEMM over K=256: one B fragment feeds two WMMAs (rows 0-15 and 16-31)
__device__ __forceinline__ AccPair gemm2_k256(float bv, const __bf16* A, const __bf16* Bp,
                                              int ntile, int lane) {
  AccPair c;
#pragma unroll
  for (int i = 0; i < 8; i++) { c.a[0][i] = bv; c.a[1][i] = bv; }
#pragma unroll
  for (int k = 0; k < 8; k++) {
    v16bf b  = loadB(Bp, ntile * 8 + k, lane);
    v16bf a0 = loadA_bf(A, LDM, k * 32, lane);
    v16bf a1 = loadA_bf(A + 16 * LDM, LDM, k * 32, lane);
    c.a[0] = wmma_bf16(a0, b, c.a[0]);
    c.a[1] = wmma_bf16(a1, b, c.a[1]);
  }
  return c;
}

__device__ __forceinline__ float sigmoidf_(float x) { return 1.f / (1.f + __expf(-x)); }

__device__ __forceinline__ float tanhf_(float x) {
  float e = __expf(-2.f * fabsf(x));
  float t = (1.f - e) / (1.f + e);
  return copysignf(t, x);
}

__global__ __launch_bounds__(32)
void tgn_edge_kernel(const float* __restrict__ memory,
                     const int* __restrict__ srcI,
                     const int* __restrict__ dstI,
                     const float* __restrict__ timev,
                     const float* __restrict__ w1,
                     const float* __restrict__ b1,
                     const float* __restrict__ b2,
                     const float* __restrict__ b_ih,
                     const float* __restrict__ b_hh,
                     const float* __restrict__ pb1,
                     const float* __restrict__ p2,
                     const float* __restrict__ pb2,
                     const __bf16* __restrict__ ws,
                     float* __restrict__ out,
                     int E) {
  // 32 edges (two 16-row tiles) per wave
  __shared__ __align__(16) __bf16 sSb[32 * LDM];     // bf16 src rows (WMMA A)
  __shared__ __align__(16) __bf16 sDb[32 * LDM];     // bf16 dst rows
  __shared__ __align__(16) __bf16 sMsg[32 * LDM];
  __shared__ __align__(16) __bf16 sNSND[64 * LDM];   // sNS | sND ; sH1 overlaid (dead by GRU)
  __bf16* sNS = sNSND;
  __bf16* sND = sNSND + 32 * LDM;
  __bf16* sH1 = sNSND;                               // 32 x LDH1 fits easily

  const int lane = threadIdx.x;
  const int col  = lane & 15;
  const int hh   = lane >> 4;
  const long e0  = (long)blockIdx.x * 32;

  const __bf16* W1p  = ws + O_W1;
  const __bf16* W2p  = ws + O_W2;
  const __bf16* WIHp = ws + O_WIH;
  const __bf16* WHHp = ws + O_WHH;
  const __bf16* P1p  = ws + O_P1;

  // ---- gather 32 src + 32 dst memory rows as bf16 panels ----
  long eg = e0 + lane;
  if (eg >= E) eg = E - 1;
  int   sid = srcI[eg];
  int   did = dstI[eg];
  float tv  = timev[eg];

  for (int r = 0; r < 32; r++) {
    int sr = __shfl(sid, r, 32);
    int dr = __shfl(did, r, 32);
    const float4* ps = (const float4*)(memory + (long)sr * 256);
    const float4* pd = (const float4*)(memory + (long)dr * 256);
#pragma unroll
    for (int c = 0; c < 2; c++) {
      int idx = lane + c * 32;           // 64 float4 per row
      *(uint2*)&sSb[r * LDM + idx * 4] = pack4bf(ps[idx]);
      *(uint2*)&sDb[r * LDM + idx * 4] = pack4bf(pd[idx]);
    }
  }

  // pre-broadcast row ids for the fp32 blend re-reads (reused for all 16 j-chunks)
  int idS[2][8], idD[2][8];
#pragma unroll
  for (int g = 0; g < 2; g++)
#pragma unroll
    for (int i = 0; i < 8; i++) {
      idS[g][i] = __shfl(sid, g * 16 + i + 8 * hh, 32);
      idD[g][i] = __shfl(did, g * 16 + i + 8 * hh, 32);
    }
  __syncthreads();

  // ---- layer1: h1 = relu([S|D]@w1[0:512] + t*w1[512] + b1)  [32x128] ----
  for (int n = 0; n < 8; n++) {
    float bv = b1[n * 16 + col];
    float wt = w1[512 * 128 + n * 16 + col];
    AccPair c;
#pragma unroll
    for (int g = 0; g < 2; g++)
#pragma unroll
      for (int i = 0; i < 8; i++) {
        float tm = __shfl(tv, g * 16 + i + 8 * hh, 32);
        c.a[g][i] = bv + tm * wt;
      }
#pragma unroll
    for (int k = 0; k < 16; k++) {
      const __bf16* P = (k < 8) ? sSb : sDb;
      int kb = (k & 7) * 32;
      v16bf b  = loadB(W1p, n * 16 + k, lane);
      v16bf a0 = loadA_bf(P, LDM, kb, lane);
      v16bf a1 = loadA_bf(P + 16 * LDM, LDM, kb, lane);
      c.a[0] = wmma_bf16(a0, b, c.a[0]);
      c.a[1] = wmma_bf16(a1, b, c.a[1]);
    }
#pragma unroll
    for (int g = 0; g < 2; g++)
#pragma unroll
      for (int i = 0; i < 8; i++)
        sH1[(g * 16 + i + 8 * hh) * LDH1 + n * 16 + col] = (__bf16)fmaxf(c.a[g][i], 0.f);
  }
  __syncthreads();

  // ---- layer2: message = h1 @ w2 + b2   [32x256] ----
  for (int n = 0; n < 16; n++) {
    float bv = b2[n * 16 + col];
    AccPair c;
#pragma unroll
    for (int i = 0; i < 8; i++) { c.a[0][i] = bv; c.a[1][i] = bv; }
#pragma unroll
    for (int k = 0; k < 4; k++) {
      v16bf b  = loadB(W2p, n * 4 + k, lane);
      v16bf a0 = loadA_bf(sH1, LDH1, k * 32, lane);
      v16bf a1 = loadA_bf(sH1 + 16 * LDH1, LDH1, k * 32, lane);
      c.a[0] = wmma_bf16(a0, b, c.a[0]);
      c.a[1] = wmma_bf16(a1, b, c.a[1]);
    }
#pragma unroll
    for (int g = 0; g < 2; g++)
#pragma unroll
      for (int i = 0; i < 8; i++)
        sMsg[(g * 16 + i + 8 * hh) * LDM + n * 16 + col] = (__bf16)c.a[g][i];
  }
  __syncthreads();

  float* outNS = out + E;
  float* outND = out + E + (long)E * 256;

  // ---- dual GRU (gi shared between src/dst), 16-column chunks, 32 rows ----
  for (int j = 0; j < 16; j++) {
    // r gate (gate 0)
    AccPair gir  = gemm2_k256(b_ih[      j * 16 + col], sMsg, WIHp,      j, lane);
    AccPair ghsr = gemm2_k256(b_hh[      j * 16 + col], sSb,  WHHp,      j, lane);
    AccPair ghdr = gemm2_k256(b_hh[      j * 16 + col], sDb,  WHHp,      j, lane);
    float rs[2][8], rd[2][8];
#pragma unroll
    for (int g = 0; g < 2; g++)
#pragma unroll
      for (int i = 0; i < 8; i++) {
        rs[g][i] = sigmoidf_(gir.a[g][i] + ghsr.a[g][i]);
        rd[g][i] = sigmoidf_(gir.a[g][i] + ghdr.a[g][i]);
      }
    // n gate (gate 2)
    AccPair gin  = gemm2_k256(b_ih[512 + j * 16 + col], sMsg, WIHp, 32 + j, lane);
    AccPair ghsn = gemm2_k256(b_hh[512 + j * 16 + col], sSb,  WHHp, 32 + j, lane);
    AccPair ghdn = gemm2_k256(b_hh[512 + j * 16 + col], sDb,  WHHp, 32 + j, lane);
    float nsv[2][8], ndv[2][8];
#pragma unroll
    for (int g = 0; g < 2; g++)
#pragma unroll
      for (int i = 0; i < 8; i++) {
        nsv[g][i] = tanhf_(gin.a[g][i] + rs[g][i] * ghsn.a[g][i]);
        ndv[g][i] = tanhf_(gin.a[g][i] + rd[g][i] * ghdn.a[g][i]);
      }
    // z gate (gate 1) + blend + store
    AccPair giz  = gemm2_k256(b_ih[256 + j * 16 + col], sMsg, WIHp, 16 + j, lane);
    AccPair ghsz = gemm2_k256(b_hh[256 + j * 16 + col], sSb,  WHHp, 16 + j, lane);
    AccPair ghdz = gemm2_k256(b_hh[256 + j * 16 + col], sDb,  WHHp, 16 + j, lane);
#pragma unroll
    for (int g = 0; g < 2; g++)
#pragma unroll
      for (int i = 0; i < 8; i++) {
        float zs = sigmoidf_(giz.a[g][i] + ghsz.a[g][i]);
        float zd = sigmoidf_(giz.a[g][i] + ghdz.a[g][i]);
        float hs = memory[(long)idS[g][i] * 256 + j * 16 + col];  // exact fp32 h (L2-hot)
        float hd = memory[(long)idD[g][i] * 256 + j * 16 + col];
        float os = (1.f - zs) * nsv[g][i] + zs * hs;
        float od = (1.f - zd) * ndv[g][i] + zd * hd;
        int m = g * 16 + i + 8 * hh;
        sNS[m * LDM + j * 16 + col] = (__bf16)os;
        sND[m * LDM + j * 16 + col] = (__bf16)od;
        long e = e0 + m;
        if (e < E) {
          outNS[e * 256 + j * 16 + col] = os;
          outND[e * 256 + j * 16 + col] = od;
        }
      }
  }
  __syncthreads();

  // ---- scorer: score = relu([NS|ND]@p1+pb1) @ p2 + pb2 ----
  float ps[2][8];
#pragma unroll
  for (int g = 0; g < 2; g++)
#pragma unroll
    for (int i = 0; i < 8; i++) ps[g][i] = 0.f;
  for (int n = 0; n < 8; n++) {
    float w  = p2[n * 16 + col];
    float bv = pb1[n * 16 + col];
    AccPair c;
#pragma unroll
    for (int i = 0; i < 8; i++) { c.a[0][i] = bv; c.a[1][i] = bv; }
#pragma unroll
    for (int k = 0; k < 16; k++) {
      const __bf16* P = (k < 8) ? sNS : sND;
      int kb = (k & 7) * 32;
      v16bf b  = loadB(P1p, n * 16 + k, lane);
      v16bf a0 = loadA_bf(P, LDM, kb, lane);
      v16bf a1 = loadA_bf(P + 16 * LDM, LDM, kb, lane);
      c.a[0] = wmma_bf16(a0, b, c.a[0]);
      c.a[1] = wmma_bf16(a1, b, c.a[1]);
    }
#pragma unroll
    for (int g = 0; g < 2; g++)
#pragma unroll
      for (int i = 0; i < 8; i++) ps[g][i] += fmaxf(c.a[g][i], 0.f) * w;
  }
#pragma unroll
  for (int off = 1; off < 16; off <<= 1) {
#pragma unroll
    for (int g = 0; g < 2; g++)
#pragma unroll
      for (int i = 0; i < 8; i++) ps[g][i] += __shfl_xor(ps[g][i], off, 32);
  }
  if (col == 0) {
    float pb2v = pb2[0];
#pragma unroll
    for (int g = 0; g < 2; g++)
#pragma unroll
      for (int i = 0; i < 8; i++) {
        long e = e0 + g * 16 + i + 8 * hh;
        if (e < E) out[e] = ps[g][i] + pb2v;
      }
  }
}

extern "C" void kernel_launch(void* const* d_in, const int* in_sizes, int n_in,
                              void* d_out, int out_size, void* d_ws, size_t ws_size,
                              hipStream_t stream) {
  const float* memory = (const float*)d_in[0];
  const int*   srcI   = (const int*)d_in[1];
  const int*   dstI   = (const int*)d_in[2];
  const float* timev  = (const float*)d_in[3];
  const float* w1     = (const float*)d_in[4];
  const float* b1     = (const float*)d_in[5];
  const float* w2     = (const float*)d_in[6];
  const float* b2     = (const float*)d_in[7];
  const float* w_ih   = (const float*)d_in[8];
  const float* w_hh   = (const float*)d_in[9];
  const float* b_ih   = (const float*)d_in[10];
  const float* b_hh   = (const float*)d_in[11];
  const float* p1     = (const float*)d_in[12];
  const float* pb1    = (const float*)d_in[13];
  const float* p2     = (const float*)d_in[14];
  const float* pb2    = (const float*)d_in[15];

  int E = in_sizes[1];
  __bf16* ws = (__bf16*)d_ws;

  // pack weights into WMMA-B fragment-major bf16 (runs every call; tiny)
  auto pack = [&](const float* W, int off, int KT, int NT, int ldw, int trans) {
    int total = KT * NT * 512;
    tgn_pack_kernel<<<(total + 255) / 256, 256, 0, stream>>>(W, ws + off, KT, NT, ldw, trans);
  };
  pack(w1,   O_W1, 16, 8, 128, 0);   // K=512 (rows 0..511), N=128
  pack(w2,   O_W2,  4, 16, 256, 0);  // K=128, N=256
  pack(w_ih, O_WIH, 8, 48, 256, 1);  // B = w_ih^T : K=256, N=768
  pack(w_hh, O_WHH, 8, 48, 256, 1);
  pack(p1,   O_P1, 16, 8, 128, 0);   // K=512, N=128

  int blocks = (E + 31) / 32;
  tgn_edge_kernel<<<blocks, 32, 0, stream>>>(memory, srcI, dstI, timev, w1, b1, b2,
                                             b_ih, b_hh, pb1, p2, pb2, ws,
                                             (float*)d_out, E);
}
